// AvaForCausalLM_30279519436983
// MI455X (gfx1250) — matrix operand order repaired and checked by reading, compiled, not run on hardware
//
#include <hip/hip_runtime.h>

// ---- model dims ----
#define L_   4
#define D_   2048
#define NH_  16
#define KVH_ 4
#define HD_  128
#define FF_  5632
#define V_   32000
#define T_   2048
#define EPS_ 1e-5f

typedef __attribute__((ext_vector_type(16))) __bf16 v16bf;
typedef __attribute__((ext_vector_type(8)))  __bf16 v8bf;
typedef __attribute__((ext_vector_type(8)))  float  v8f;
typedef __attribute__((ext_vector_type(4)))  float  v4f;
typedef __attribute__((ext_vector_type(4)))  int    v4i;

#if __has_builtin(__builtin_amdgcn_global_load_async_to_lds_b128)
#define ASYNC_LDS 1
// builtin signature: (int4 AS1* gsrc, int4 AS3* ldst, imm int offset, imm int cpol)
#define GV4(p) ((__attribute__((address_space(1))) v4i*)(p))
#define LV4(p) ((__attribute__((address_space(3))) v4i*)(p))
#else
#define ASYNC_LDS 0
#endif

// f32 -> bf16, round-to-nearest (cheap: one add + shift)
static __device__ __forceinline__ __bf16 f2bf(float f) {
  unsigned u = __builtin_bit_cast(unsigned, f);
  unsigned short hh = (unsigned short)((u + 0x8000u) >> 16);
  return __builtin_bit_cast(__bf16, hh);
}

static __device__ __forceinline__ void wait_async0() {
#if ASYNC_LDS
#if __has_builtin(__builtin_amdgcn_s_wait_asynccnt)
  __builtin_amdgcn_s_wait_asynccnt(0);
#else
  asm volatile("s_wait_asynccnt 0x0" ::: "memory");
#endif
#endif
}

// ---------------- embedding gather ----------------
__global__ void k_embed(const int* __restrict__ ids, const float* __restrict__ emb,
                        float* __restrict__ h) {
  int t = blockIdx.x;
  int tok = ids[t];
  const float* src = emb + (size_t)tok * D_;
  float* dst = h + (size_t)t * D_;
  for (int d = threadIdx.x; d < D_; d += blockDim.x) dst[d] = src[d];
}

// ---------------- RMSNorm -> bf16 activations ----------------
__global__ __launch_bounds__(256) void k_rms_bf16(const float* __restrict__ h,
                                                  const float* __restrict__ w,
                                                  __bf16* __restrict__ out) {
  __shared__ float red[256];
  int t = blockIdx.x;
  const float* row = h + (size_t)t * D_;
  float s = 0.f;
  for (int d = threadIdx.x; d < D_; d += 256) { float v = row[d]; s += v * v; }
  red[threadIdx.x] = s;
  __syncthreads();
  for (int off = 128; off > 0; off >>= 1) {
    if (threadIdx.x < off) red[threadIdx.x] += red[threadIdx.x + off];
    __syncthreads();
  }
  float rstd = rsqrtf(red[0] * (1.0f / D_) + EPS_);
  __bf16* o = out + (size_t)t * D_;
  for (int d = threadIdx.x; d < D_; d += 256) o[d] = f2bf(row[d] * rstd * w[d]);
}

// ---------------- WMMA GEMM: C[M,N](f32) = A[M,K](bf16) * B[K,N](f32->bf16) ----------------
#define BM 128
#define BN 128
#define BK 64   // two WMMA K-steps per LDS stage -> 16 wmma per barrier

__global__ __launch_bounds__(256) void k_gemm(const __bf16* __restrict__ A,
                                              const float* __restrict__ B,
                                              float* __restrict__ C,
                                              int M, int N, int K, int accFlag) {
  __shared__ __align__(32) __bf16 As[BM][BK];   // row-major [m][k]
  __shared__ __align__(32) __bf16 Bs[BN][BK];   // transposed [n][k]
  int tid  = threadIdx.x;
  int lane = tid & 31;
  int wid  = tid >> 5;
  int wm = wid & 3;          // 4 waves along M (32 rows each)
  int wn = wid >> 2;         // 2 waves along N (64 cols each)
  int mBase = blockIdx.y * BM, nBase = blockIdx.x * BN;
  int half = lane >> 4, l15 = lane & 15;

  v8f zero = {0.f, 0.f, 0.f, 0.f, 0.f, 0.f, 0.f, 0.f};
  v8f acc[2][4];
  for (int i = 0; i < 2; i++)
    for (int j = 0; j < 4; j++) acc[i][j] = zero;

  int ar  = tid >> 1, ac = (tid & 1) * 32;   // A staging: 2 threads/row, 32 bf16 (64B) each
  int bkr = tid >> 3, bc = (tid & 7) * 16;   // B staging: 8 threads/row, 16 f32 each, 2 row passes

  for (int k0 = 0; k0 < K; k0 += BK) {
    // ---- stage A (bf16, raw copy -> async to LDS when available) ----
    const __bf16* agp = A + (size_t)(mBase + ar) * K + k0 + ac;
#if ASYNC_LDS
#pragma unroll
    for (int x = 0; x < 4; x++)
      __builtin_amdgcn_global_load_async_to_lds_b128(
          GV4(agp + x * 8), LV4(&As[ar][ac + x * 8]), 0, 0);
#else
#pragma unroll
    for (int x = 0; x < 4; x++)
      *(v8bf*)&As[ar][ac + x * 8] = *(const v8bf*)(agp + x * 8);
#endif
    // ---- stage B: f32 load, convert to bf16, store transposed [n][k] ----
#pragma unroll
    for (int p = 0; p < 2; p++) {
      int krow = bkr + p * 32;
      const float* bp = B + (size_t)(k0 + krow) * N + nBase + bc;
      v4f b0 = *(const v4f*)(bp + 0);
      v4f b1 = *(const v4f*)(bp + 4);
      v4f b2 = *(const v4f*)(bp + 8);
      v4f b3 = *(const v4f*)(bp + 12);
      float tmp[16];
#pragma unroll
      for (int j = 0; j < 4; j++) { tmp[j] = b0[j]; tmp[4+j] = b1[j]; tmp[8+j] = b2[j]; tmp[12+j] = b3[j]; }
#pragma unroll
      for (int j = 0; j < 16; j++) Bs[bc + j][krow] = f2bf(tmp[j]);
    }
    if (k0 + BK < K)
      __builtin_prefetch(B + (size_t)(k0 + BK + bkr) * N + nBase + bc, 0, 1);
    wait_async0();
    __syncthreads();

    // ---- 2 K-substeps x 8 WMMA ----
#pragma unroll
    for (int ks = 0; ks < 2; ks++) {
      int ko = ks * 32;
      v16bf a[2], b[4];
#pragma unroll
      for (int mi = 0; mi < 2; mi++) {
        int arow = wm * 32 + mi * 16 + l15;
        v8bf lo = *(const v8bf*)&As[arow][ko + half * 8];
        v8bf hi = *(const v8bf*)&As[arow][ko + 16 + half * 8];
        a[mi] = __builtin_shufflevector(lo, hi, 0,1,2,3,4,5,6,7,8,9,10,11,12,13,14,15);
      }
#pragma unroll
      for (int ni = 0; ni < 4; ni++) {
        int ncol = wn * 64 + ni * 16 + l15;
        v8bf lo = *(const v8bf*)&Bs[ncol][ko + half * 8];
        v8bf hi = *(const v8bf*)&Bs[ncol][ko + 16 + half * 8];
        b[ni] = __builtin_shufflevector(lo, hi, 0,1,2,3,4,5,6,7,8,9,10,11,12,13,14,15);
      }
#pragma unroll
      for (int mi = 0; mi < 2; mi++)
#pragma unroll
        for (int ni = 0; ni < 4; ni++)
          acc[mi][ni] = __builtin_amdgcn_wmma_f32_16x16x32_bf16(
              false, a[mi], false, b[ni], (short)0, acc[mi][ni], false, false);
    }
    __syncthreads();
  }

  // epilogue: C layout — VGPR r, lane l: M = r + 8*(l>=16), N = l&15
#pragma unroll
  for (int mi = 0; mi < 2; mi++)
#pragma unroll
    for (int ni = 0; ni < 4; ni++) {
      int col = nBase + wn * 64 + ni * 16 + l15;
#pragma unroll
      for (int r = 0; r < 8; r++) {
        int row = mBase + wm * 32 + mi * 16 + r + 8 * half;
        size_t idx = (size_t)row * N + col;
        float vv = acc[mi][ni][r];
        if (accFlag) C[idx] += vv; else C[idx] = vv;
      }
    }
}

// ---------------- RoPE (in place) ----------------
__global__ void k_rope(float* __restrict__ p, int H, int n) {
  int i = blockIdx.x * blockDim.x + threadIdx.x;
  if (i >= n) return;
  int j = i & 63;          // freq index 0..63
  int r = i >> 6;
  int h = r % H;
  int t = r / H;
  float e = (float)(2 * j) * (1.0f / HD_);
  float inv = __powf(10000.0f, -e);
  float th = (float)t * inv;
  float c = __cosf(th), s = __sinf(th);
  float* base = p + (size_t)t * ((size_t)H * HD_) + (size_t)h * HD_;
  float x0 = base[j], x1 = base[j + 64];
  base[j]      = x0 * c - x1 * s;
  base[j + 64] = x1 * c + x0 * s;
}

// ---------------- WMMA flash attention (GQA, causal), bf16 out ----------------
// grid (T/64, NH), 128 threads = 4 waves; each wave owns 16 query rows.
#define AQ 64
__global__ __launch_bounds__(128) void k_attn(const float* __restrict__ q,
                                              const float* __restrict__ k,
                                              const float* __restrict__ v,
                                              const int* __restrict__ am,
                                              __bf16* __restrict__ out) {
  __shared__ __align__(32) __bf16 Qs[AQ][HD_];   // [q][d]   16KB
  __shared__ __align__(32) __bf16 Ks[32][HD_];   // [key][d]  8KB
  __shared__ __align__(32) __bf16 Vs[HD_][32];   // [d][key]  8KB
  __shared__ __align__(32) __bf16 Ps[4][16][32]; // per-wave P 4KB
  int tid = threadIdx.x, lane = tid & 31, wv = tid >> 5;
  int half = lane >> 4, l15 = lane & 15;
  int hh = blockIdx.y, kvh = hh / (NH_ / KVH_);
  int t0 = blockIdx.x * AQ;
  const float scale = 0.0883883476483f; // 1/sqrt(128)

  // stage Q (scaled) as bf16
  {
    int r = tid >> 1, c0 = (tid & 1) * 64;
    const float* qp = q + (size_t)(t0 + r) * (NH_ * HD_) + hh * HD_ + c0;
#pragma unroll 8
    for (int j = 0; j < 64; j++) Qs[r][c0 + j] = f2bf(qp[j] * scale);
  }
  __syncthreads();

  v8f zero = {0.f, 0.f, 0.f, 0.f, 0.f, 0.f, 0.f, 0.f};
  v8f ctx[8];
#pragma unroll
  for (int i = 0; i < 8; i++) ctx[i] = zero;
  float mrow[8], lrow[8];
#pragma unroll
  for (int r = 0; r < 8; r++) { mrow[r] = -1e30f; lrow[r] = 0.f; }

  int nk = t0 + AQ;          // causal: keys needed by this block (multiple of 32)
  for (int kb = 0; kb < nk; kb += 32) {
    // stage 32-key K/V tile (f32 -> bf16); Vs transposed
    {
      int kr = tid >> 2, c0 = (tid & 3) * 32;
      const float* kp = k + (size_t)(kb + kr) * (KVH_ * HD_) + kvh * HD_ + c0;
      const float* vp = v + (size_t)(kb + kr) * (KVH_ * HD_) + kvh * HD_ + c0;
#pragma unroll 8
      for (int j = 0; j < 32; j++) Ks[kr][c0 + j] = f2bf(kp[j]);
#pragma unroll 8
      for (int j = 0; j < 32; j++) Vs[c0 + j][kr] = f2bf(vp[j]);
    }
    __syncthreads();

    // S = Q(16x128) x K^T(128x32): 2 score tiles x 4 K-substeps
    v8f s[2] = {zero, zero};
#pragma unroll
    for (int kk = 0; kk < 4; kk++) {
      int qrow = wv * 16 + l15;
      v8bf alo = *(const v8bf*)&Qs[qrow][kk * 32 + half * 8];
      v8bf ahi = *(const v8bf*)&Qs[qrow][kk * 32 + 16 + half * 8];
      v16bf a = __builtin_shufflevector(alo, ahi, 0,1,2,3,4,5,6,7,8,9,10,11,12,13,14,15);
#pragma unroll
      for (int ni = 0; ni < 2; ni++) {
        v8bf blo = *(const v8bf*)&Ks[ni * 16 + l15][kk * 32 + half * 8];
        v8bf bhi = *(const v8bf*)&Ks[ni * 16 + l15][kk * 32 + 16 + half * 8];
        v16bf b = __builtin_shufflevector(blo, bhi, 0,1,2,3,4,5,6,7,8,9,10,11,12,13,14,15);
        s[ni] = __builtin_amdgcn_wmma_f32_16x16x32_bf16(
            false, a, false, b, (short)0, s[ni], false, false);
      }
    }

    // mask (causal + attention_mask) in C layout
#pragma unroll
    for (int ni = 0; ni < 2; ni++) {
      int key = kb + ni * 16 + l15;
      float madd = (1.0f - (float)am[key]) * -3.0e38f;
#pragma unroll
      for (int r = 0; r < 8; r++) {
        int qg = t0 + wv * 16 + r + 8 * half;
        float sv = s[ni][r] + madd;
        if (key > qg) sv = -1e30f;
        s[ni][r] = sv;
      }
    }

    // online softmax per row; P -> LDS (bf16, A-layout staging)
#pragma unroll
    for (int r = 0; r < 8; r++) {
      float rm = fmaxf(s[0][r], s[1][r]);
#pragma unroll
      for (int msk = 1; msk < 16; msk <<= 1) rm = fmaxf(rm, __shfl_xor(rm, msk, 16));
      float mn = fmaxf(mrow[r], rm);
      float cor = __expf(mrow[r] - mn);
      float e0 = (s[0][r] <= -1e29f) ? 0.f : __expf(s[0][r] - mn);
      float e1 = (s[1][r] <= -1e29f) ? 0.f : __expf(s[1][r] - mn);
      float rs = e0 + e1;
#pragma unroll
      for (int msk = 1; msk < 16; msk <<= 1) rs += __shfl_xor(rs, msk, 16);
      lrow[r] = lrow[r] * cor + rs;
      mrow[r] = mn;
#pragma unroll
      for (int ni = 0; ni < 8; ni++) ctx[ni][r] *= cor;
      Ps[wv][r + 8 * half][l15]      = f2bf(e0);
      Ps[wv][r + 8 * half][16 + l15] = f2bf(e1);
    }
    // wave-private LDS RAW: make P stores visible before fragment reads
    asm volatile("s_wait_dscnt 0x0" ::: "memory");

    // ctx += P(16x32) x V(32x128): 8 output tiles
    {
      v8bf plo = *(const v8bf*)&Ps[wv][l15][half * 8];
      v8bf phi = *(const v8bf*)&Ps[wv][l15][16 + half * 8];
      v16bf a = __builtin_shufflevector(plo, phi, 0,1,2,3,4,5,6,7,8,9,10,11,12,13,14,15);
#pragma unroll
      for (int ni = 0; ni < 8; ni++) {
        v8bf blo = *(const v8bf*)&Vs[ni * 16 + l15][half * 8];
        v8bf bhi = *(const v8bf*)&Vs[ni * 16 + l15][16 + half * 8];
        v16bf b = __builtin_shufflevector(blo, bhi, 0,1,2,3,4,5,6,7,8,9,10,11,12,13,14,15);
        ctx[ni] = __builtin_amdgcn_wmma_f32_16x16x32_bf16(
            false, a, false, b, (short)0, ctx[ni], false, false);
      }
    }
    __syncthreads();  // before next tile overwrites Ks/Vs
  }

  // epilogue: normalize by row sum, emit bf16 [t][h*128 + d]
#pragma unroll
  for (int ni = 0; ni < 8; ni++) {
#pragma unroll
    for (int r = 0; r < 8; r++) {
      int qg = t0 + wv * 16 + r + 8 * half;
      out[(size_t)qg * (NH_ * HD_) + hh * HD_ + ni * 16 + l15] = f2bf(ctx[ni][r] / lrow[r]);
    }
  }
}

// ---------------- SiLU gate -> bf16 ----------------
__global__ void k_silu_gate(const float* __restrict__ g, const float* __restrict__ u,
                            __bf16* __restrict__ gu, int n) {
  int i = blockIdx.x * blockDim.x + threadIdx.x;
  if (i >= n) return;
  float gv = g[i];
  float sv = gv / (1.0f + __expf(-gv));
  gu[i] = f2bf(sv * u[i]);
}

// ---------------- launcher ----------------
extern "C" void kernel_launch(void* const* d_in, const int* in_sizes, int n_in,
                              void* d_out, int out_size, void* d_ws, size_t ws_size,
                              hipStream_t stream) {
  (void)in_sizes; (void)n_in; (void)out_size; (void)ws_size;
  const int*   ids   = (const int*)d_in[0];
  const int*   amask = (const int*)d_in[1];
  const float* emb   = (const float*)d_in[2];
  const float* Wq    = (const float*)d_in[3];
  const float* Wk    = (const float*)d_in[4];
  const float* Wv    = (const float*)d_in[5];
  const float* Wo    = (const float*)d_in[6];
  const float* ln1   = (const float*)d_in[7];
  const float* ln2   = (const float*)d_in[8];
  const float* Wg    = (const float*)d_in[9];
  const float* Wu    = (const float*)d_in[10];
  const float* Wd    = (const float*)d_in[11];
  const float* normw = (const float*)d_in[12];
  const float* lm    = (const float*)d_in[13];
  float* out = (float*)d_out;

  char* wp = (char*)d_ws;
  auto carve = [&](size_t bytes) -> void* {
    void* p = (void*)wp;
    wp += (bytes + 255) & ~(size_t)255;
    return p;
  };
  float*  h   = (float*) carve((size_t)T_ * D_ * 4);
  __bf16* xb  = (__bf16*)carve((size_t)T_ * D_ * 2);
  float*  q   = (float*) carve((size_t)T_ * NH_ * HD_ * 4);
  float*  kk  = (float*) carve((size_t)T_ * KVH_ * HD_ * 4);
  float*  vv  = (float*) carve((size_t)T_ * KVH_ * HD_ * 4);
  __bf16* cb  = (__bf16*)carve((size_t)T_ * D_ * 2);
  float*  g   = (float*) carve((size_t)T_ * FF_ * 4);
  float*  u   = (float*) carve((size_t)T_ * FF_ * 4);
  __bf16* gu  = (__bf16*)carve((size_t)T_ * FF_ * 2);

  k_embed<<<T_, 256, 0, stream>>>(ids, emb, h);

  for (int i = 0; i < L_; i++) {
    k_rms_bf16<<<T_, 256, 0, stream>>>(h, ln1 + (size_t)i * D_, xb);
    k_gemm<<<dim3((NH_*HD_)/BN,  T_/BM), 256, 0, stream>>>(xb, Wq + (size_t)i * D_ * (NH_*HD_),  q,  T_, NH_*HD_,  D_, 0);
    k_gemm<<<dim3((KVH_*HD_)/BN, T_/BM), 256, 0, stream>>>(xb, Wk + (size_t)i * D_ * (KVH_*HD_), kk, T_, KVH_*HD_, D_, 0);
    k_gemm<<<dim3((KVH_*HD_)/BN, T_/BM), 256, 0, stream>>>(xb, Wv + (size_t)i * D_ * (KVH_*HD_), vv, T_, KVH_*HD_, D_, 0);
    k_rope<<<(T_*NH_*64)/256,  256, 0, stream>>>(q,  NH_,  T_*NH_*64);
    k_rope<<<(T_*KVH_*64)/256, 256, 0, stream>>>(kk, KVH_, T_*KVH_*64);
    k_attn<<<dim3(T_/AQ, NH_), 128, 0, stream>>>(q, kk, vv, amask, cb);
    k_gemm<<<dim3(D_/BN, T_/BM), 256, 0, stream>>>(cb, Wo + (size_t)i * D_ * D_, h, T_, D_, D_, 1);
    k_rms_bf16<<<T_, 256, 0, stream>>>(h, ln2 + (size_t)i * D_, xb);
    k_gemm<<<dim3(FF_/BN, T_/BM), 256, 0, stream>>>(xb, Wg + (size_t)i * D_ * FF_, g, T_, FF_, D_, 0);
    k_gemm<<<dim3(FF_/BN, T_/BM), 256, 0, stream>>>(xb, Wu + (size_t)i * D_ * FF_, u, T_, FF_, D_, 0);
    k_silu_gate<<<(T_*FF_)/256, 256, 0, stream>>>(g, u, gu, T_*FF_);
    k_gemm<<<dim3(D_/BN, T_/BM), 256, 0, stream>>>(gu, Wd + (size_t)i * FF_ * D_, h, T_, D_, FF_, 1);
  }

  k_rms_bf16<<<T_, 256, 0, stream>>>(h, normw, xb);
  k_gemm<<<dim3(V_/BN, T_/BM), 256, 0, stream>>>(xb, lm, out, T_, V_, D_, 0);
}
